// MultiHeadAttention_48739288875409
// MI455X (gfx1250) — matrix operand order repaired and checked by reading, not compiled, not run on hardware
//
#include <hip/hip_runtime.h>

#define S_LEN   4096
#define DM      768
#define NH      12
#define DH      64

typedef __attribute__((ext_vector_type(16))) __bf16       v16bf;
typedef __attribute__((ext_vector_type(8)))  float        v8f;
typedef __attribute__((ext_vector_type(4)))  unsigned int u32x4;
typedef __attribute__((ext_vector_type(4)))  float        f32x4;

union BF16Frag {
  v16bf  v;
  __bf16 h[16];
  u32x4  q[2];
};

__device__ __forceinline__ unsigned short f32_to_bf16_bits(float f) {
  __bf16 t = (__bf16)f;
  return __builtin_bit_cast(unsigned short, t);
}
__device__ __forceinline__ unsigned int pack_bf16x2(float a, float b) {
  union { unsigned int u; __bf16 h[2]; } r;
  r.h[0] = (__bf16)a; r.h[1] = (__bf16)b;
  return r.u;
}

// ---------------------------------------------------------------------------
// Prepass: f32 -> bf16 bulk convert (8 elements / thread)
// ---------------------------------------------------------------------------
__global__ __launch_bounds__(256) void cvt_bf16_kernel(
    const float* __restrict__ src, unsigned short* __restrict__ dst, int n8)
{
  const int i = blockIdx.x * 256 + threadIdx.x;
  if (i >= n8) return;
  const f32x4 lo = ((const f32x4*)src)[2 * i];
  const f32x4 hi = ((const f32x4*)src)[2 * i + 1];
  union { u32x4 q; __bf16 h[8]; } r;
  #pragma unroll
  for (int e = 0; e < 4; ++e) {
    r.h[e]     = (__bf16)lo[e];
    r.h[e + 4] = (__bf16)hi[e];
  }
  ((u32x4*)dst)[i] = r.q;
}

// ---------------------------------------------------------------------------
// Shared GEMM fragment loaders (32-bit offsets; uniform base + lane offset)
// ---------------------------------------------------------------------------
__device__ __forceinline__ void load_a_frag(const unsigned short* __restrict__ p,
                                            int off, BF16Frag& a) {
  a.q[0] = ((const u32x4*)(p + off))[0];
  a.q[1] = ((const u32x4*)(p + off))[1];
}
__device__ __forceinline__ void load_b_frag(const unsigned short* __restrict__ p,
                                            int off, BF16Frag& b) {
  b.q[0] = ((const u32x4*)(p + off))[0];
  b.q[1] = ((const u32x4*)(p + off))[1];
}

// ---------------------------------------------------------------------------
// Kernel 1: qkv = x @ W_in^T + b_in -> Q[H,S,Dh] (pre-scaled by 1/8),
// K[H,S,Dh], Vt[H,Dh,S] (all bf16). One wave: 16x64 tile of [4096 x 2304].
// Software-pipelined: next k-slice fragments prefetched during WMMAs.
// ---------------------------------------------------------------------------
__global__ __launch_bounds__(128) void qkv_gemm_kernel(
    const unsigned short* __restrict__ Xb, const unsigned short* __restrict__ Wb,
    const float* __restrict__ bias,
    unsigned short* __restrict__ Qb, unsigned short* __restrict__ Kb,
    unsigned short* __restrict__ Vt)
{
  const int lane  = threadIdx.x & 31;
  const int l16   = lane & 15;
  const int lhalf = lane >> 4;
  const int wid   = blockIdx.x * (blockDim.x >> 5) + (threadIdx.x >> 5);
  const int mTile = wid & 255;
  const int nGrp  = wid >> 8;
  const int mBase = mTile * 16;
  const int nBase = nGrp * 64;

  const int aOff = (mBase + l16) * DM + 16 * lhalf;
  int wOff[4];
  #pragma unroll
  for (int nb = 0; nb < 4; ++nb)
    wOff[nb] = (nBase + nb * 16 + l16) * DM + 16 * lhalf;

  v8f acc[4] = {};
  BF16Frag a, b[4];
  load_a_frag(Xb, aOff, a);
  #pragma unroll
  for (int nb = 0; nb < 4; ++nb) load_b_frag(Wb, wOff[nb], b[nb]);

  #pragma unroll 2
  for (int kb = 0; kb < DM - 32; kb += 32) {
    BF16Frag an, bn[4];
    load_a_frag(Xb, aOff + kb + 32, an);
    #pragma unroll
    for (int nb = 0; nb < 4; ++nb) load_b_frag(Wb, wOff[nb] + kb + 32, bn[nb]);
    #pragma unroll
    for (int nb = 0; nb < 4; ++nb)
      acc[nb] = __builtin_amdgcn_wmma_f32_16x16x32_bf16(
          false, a.v, false, b[nb].v, (short)0, acc[nb], false, false);
    a = an;
    #pragma unroll
    for (int nb = 0; nb < 4; ++nb) b[nb] = bn[nb];
  }
  #pragma unroll
  for (int nb = 0; nb < 4; ++nb)
    acc[nb] = __builtin_amdgcn_wmma_f32_16x16x32_bf16(
        false, a.v, false, b[nb].v, (short)0, acc[nb], false, false);

  #pragma unroll
  for (int nb = 0; nb < 4; ++nb) {
    const int blk     = nBase + nb * 16;      // uniform -> SALU div/mod
    const int section = blk / DM;             // 0=Q 1=K 2=V
    const int rem     = blk - section * DM;
    const int h       = rem >> 6;
    const int dh      = (rem & 63) + l16;
    const float bv    = bias[blk + l16];
    #pragma unroll
    for (int r = 0; r < 8; ++r) {
      const int row = mBase + r + 8 * lhalf;
      const float val = acc[nb][r] + bv;
      if (section == 0)      Qb[(h * S_LEN + row) * DH + dh] = f32_to_bf16_bits(val * 0.125f);
      else if (section == 1) Kb[(h * S_LEN + row) * DH + dh] = f32_to_bf16_bits(val);
      else                   Vt[(h * DH + dh) * S_LEN + row] = f32_to_bf16_bits(val);
    }
  }
}

// ---------------------------------------------------------------------------
// Kernel 2: causal flash attention, transposed formulation:
//   S^T = K * Q^T  (rows = kv, cols = q)   ->  per-lane softmax stats
//   O^T = V^T * P  (rows = dh, cols = q)
// One wave per (head, 16-query tile); KV chunks of 32; K fragments for the
// next chunk are prefetched while softmax/LDS/PV of the current chunk runs.
// ---------------------------------------------------------------------------
__device__ __forceinline__ void load_ka(const unsigned short* __restrict__ Kb,
                                        int h, int kvBase, int l16, int lhalf,
                                        BF16Frag (&ka)[4])
{
  const unsigned short* kr0 = Kb + (h * S_LEN + kvBase + l16) * DH + 8 * lhalf;
  const unsigned short* kr1 = kr0 + 16 * DH;
  ka[0].q[0] = ((const u32x4*)kr0)[0]; ka[0].q[1] = ((const u32x4*)kr0)[2];
  ka[1].q[0] = ((const u32x4*)kr1)[0]; ka[1].q[1] = ((const u32x4*)kr1)[2];
  ka[2].q[0] = ((const u32x4*)kr0)[4]; ka[2].q[1] = ((const u32x4*)kr0)[6];
  ka[3].q[0] = ((const u32x4*)kr1)[4]; ka[3].q[1] = ((const u32x4*)kr1)[6];
}

template <bool MASKED>
__device__ __forceinline__ void attn_chunk(
    int kvBase, int qBase, int h, int l16, int lhalf,
    const unsigned short* __restrict__ Kb, const unsigned short* __restrict__ Vt,
    unsigned short* ps, const BF16Frag (&qB)[2], BF16Frag (&ka)[4],
    v8f (&o)[4], float& mrow, float& lpart)
{
  // ---- 4 score WMMAs from preloaded K fragments ----
  v8f st0 = {}, st1 = {};
  st0 = __builtin_amdgcn_wmma_f32_16x16x32_bf16(false, ka[0].v, false, qB[0].v, (short)0, st0, false, false);
  st1 = __builtin_amdgcn_wmma_f32_16x16x32_bf16(false, ka[1].v, false, qB[0].v, (short)0, st1, false, false);
  st0 = __builtin_amdgcn_wmma_f32_16x16x32_bf16(false, ka[2].v, false, qB[1].v, (short)0, st0, false, false);
  st1 = __builtin_amdgcn_wmma_f32_16x16x32_bf16(false, ka[3].v, false, qB[1].v, (short)0, st1, false, false);

  // ---- V^T fragments + next chunk's K fragments: in flight under softmax ----
  BF16Frag va[4];
  #pragma unroll
  for (int nb = 0; nb < 4; ++nb) {
    const unsigned short* vrow =
        Vt + (h * DH + nb * 16 + l16) * S_LEN + kvBase + 8 * lhalf;
    va[nb].q[0] = ((const u32x4*)vrow)[0];
    va[nb].q[1] = ((const u32x4*)vrow)[2];
  }
  BF16Frag kan[4];
  if (!MASKED) load_ka(Kb, h, kvBase + 32, l16, lhalf, kan);

  // ---- per-lane online softmax (lane owns q = qBase + l16) ----
  const int qi = qBase + l16;
  float mx = mrow;
  #pragma unroll
  for (int r = 0; r < 8; ++r) {
    float a = st0[r], b = st1[r];
    if (MASKED) {
      const int kv0 = kvBase + r + 8 * lhalf;
      a = (kv0      <= qi) ? a : -__builtin_inff();
      b = (kv0 + 16 <= qi) ? b : -__builtin_inff();
      st0[r] = a; st1[r] = b;
    }
    mx = fmaxf(mx, fmaxf(a, b));
  }
  mx = fmaxf(mx, __shfl_xor(mx, 16));   // combine kv halves (lane +-16)
  const float corr = __expf(mrow - mx);
  mrow = mx;

  float sum = 0.0f;
  union { u32x4 q; __bf16 hh[8]; } p0, p1;
  #pragma unroll
  for (int r = 0; r < 8; ++r) {
    const float e0 = __expf(st0[r] - mx);
    const float e1 = __expf(st1[r] - mx);
    sum += e0 + e1;
    p0.hh[r] = (__bf16)e0;
    p1.hh[r] = (__bf16)e1;
  }
  lpart = lpart * corr + sum;
  #pragma unroll
  for (int nb = 0; nb < 4; ++nb)
    #pragma unroll
    for (int r = 0; r < 8; ++r) o[nb][r] *= corr;

  // ---- P (32kv x 16q) through LDS: row q, contiguous kv per lane ----
  *(u32x4*)(ps + l16 * 32 + 8 * lhalf)      = p0.q;   // kv = 8*lhalf + [0..7]
  *(u32x4*)(ps + l16 * 32 + 16 + 8 * lhalf) = p1.q;   // kv = 16 + 8*lhalf + [0..7]
  asm volatile("s_wait_dscnt 0" ::: "memory");
  BF16Frag pB;   // B-layout: lane l16 = q, elements kv = 8*lhalf + e (e<8), 16 + 8*lhalf + (e-8) (e>=8)
  {
    const u32x4* pr = (const u32x4*)(ps + l16 * 32 + 8 * lhalf);
    pB.q[0] = pr[0]; pB.q[1] = pr[2];
  }
  asm volatile("" ::: "memory");

  // ---- O^T += V^T * P ----
  #pragma unroll
  for (int nb = 0; nb < 4; ++nb)
    o[nb] = __builtin_amdgcn_wmma_f32_16x16x32_bf16(
        false, va[nb].v, false, pB.v, (short)0, o[nb], false, false);

  if (!MASKED) {
    #pragma unroll
    for (int t = 0; t < 4; ++t) ka[t] = kan[t];   // SSA rotation
  }
}

__global__ __launch_bounds__(128) void attn_kernel(
    const unsigned short* __restrict__ Qb,
    const unsigned short* __restrict__ Kb,
    const unsigned short* __restrict__ Vt,
    unsigned short* __restrict__ Ob)
{
  __shared__ unsigned short pSh[4 * 16 * 32];

  const int lane  = threadIdx.x & 31;
  const int l16   = lane & 15;
  const int lhalf = lane >> 4;
  const int wiv   = threadIdx.x >> 5;
  const int wid   = blockIdx.x * 4 + wiv;
  const int h     = wid >> 8;
  const int i     = wid & 255;
  // zig-zag pairing: adjacent waves get complementary causal work
  const int qt    = (i & 1) ? (255 - (i >> 1)) : (i >> 1);
  const int qBase = qt * 16;

  unsigned short* ps = pSh + wiv * (16 * 32);

  // Q^T as B-fragments: lane l16 = q col; elements dh = 32*c + 8*lhalf + e (e<8), 32*c + 16 + 8*lhalf + (e-8) (e>=8)
  BF16Frag qB[2];
  {
    const unsigned short* qrow =
        Qb + (h * S_LEN + qBase + l16) * DH + 8 * lhalf;
    qB[0].q[0] = ((const u32x4*)qrow)[0]; qB[0].q[1] = ((const u32x4*)qrow)[2];
    qB[1].q[0] = ((const u32x4*)qrow)[4]; qB[1].q[1] = ((const u32x4*)qrow)[6];
  }

  v8f o[4] = {};
  float mrow = -__builtin_inff();
  float lpart = 0.0f;

  const int lastBase = ((qBase + 15) >> 5) << 5;   // only chunk crossing diagonal
  BF16Frag ka[4];
  load_ka(Kb, h, 0, l16, lhalf, ka);
  for (int kvBase = 0; kvBase < lastBase; kvBase += 32)
    attn_chunk<false>(kvBase, qBase, h, l16, lhalf, Kb, Vt, ps, qB, ka, o, mrow, lpart);
  attn_chunk<true>(lastBase, qBase, h, l16, lhalf, Kb, Vt, ps, qB, ka, o, mrow, lpart);

  // final: l per q = sum over both kv halves
  const float l = lpart + __shfl_xor(lpart, 16);
  const float linv = __builtin_amdgcn_rcpf(l);

  const int rowOff = (qBase + l16) * DM;
  #pragma unroll
  for (int nb = 0; nb < 4; ++nb) {
    #pragma unroll
    for (int rp = 0; rp < 4; ++rp) {
      const unsigned int pk =
          pack_bf16x2(o[nb][2 * rp] * linv, o[nb][2 * rp + 1] * linv);
      const int col = h * DH + nb * 16 + 8 * lhalf + 2 * rp;
      *(unsigned int*)(Ob + rowOff + col) = pk;
    }
  }
}

// ---------------------------------------------------------------------------
// Kernel 3: out = O @ W_out^T + b_out  (f32 output), both operands bf16
// ---------------------------------------------------------------------------
__global__ __launch_bounds__(128) void out_gemm_kernel(
    const unsigned short* __restrict__ Ob, const unsigned short* __restrict__ Wb,
    const float* __restrict__ bias, float* __restrict__ Out)
{
  const int lane  = threadIdx.x & 31;
  const int l16   = lane & 15;
  const int lhalf = lane >> 4;
  const int wid   = blockIdx.x * (blockDim.x >> 5) + (threadIdx.x >> 5);
  const int mTile = wid & 255;
  const int nGrp  = wid >> 8;
  const int mBase = mTile * 16;
  const int nBase = nGrp * 64;

  const int aOff = (mBase + l16) * DM + 16 * lhalf;
  int wOff[4];
  #pragma unroll
  for (int nb = 0; nb < 4; ++nb)
    wOff[nb] = (nBase + nb * 16 + l16) * DM + 16 * lhalf;

  v8f acc[4] = {};
  BF16Frag a, b[4];
  load_a_frag(Ob, aOff, a);
  #pragma unroll
  for (int nb = 0; nb < 4; ++nb) load_b_frag(Wb, wOff[nb], b[nb]);

  #pragma unroll 2
  for (int kb = 0; kb < DM - 32; kb += 32) {
    BF16Frag an, bn[4];
    load_a_frag(Ob, aOff + kb + 32, an);
    #pragma unroll
    for (int nb = 0; nb < 4; ++nb) load_b_frag(Wb, wOff[nb] + kb + 32, bn[nb]);
    #pragma unroll
    for (int nb = 0; nb < 4; ++nb)
      acc[nb] = __builtin_amdgcn_wmma_f32_16x16x32_bf16(
          false, a.v, false, b[nb].v, (short)0, acc[nb], false, false);
    a = an;
    #pragma unroll
    for (int nb = 0; nb < 4; ++nb) b[nb] = bn[nb];
  }
  #pragma unroll
  for (int nb = 0; nb < 4; ++nb)
    acc[nb] = __builtin_amdgcn_wmma_f32_16x16x32_bf16(
        false, a.v, false, b[nb].v, (short)0, acc[nb], false, false);

  #pragma unroll
  for (int nb = 0; nb < 4; ++nb) {
    const int col = nBase + nb * 16 + l16;
    const float bv = bias[col];
    #pragma unroll
    for (int r = 0; r < 8; ++r) {
      const int row = mBase + r + 8 * lhalf;
      Out[row * DM + col] = acc[nb][r] + bv;
    }
  }
}

// ---------------------------------------------------------------------------
extern "C" void kernel_launch(void* const* d_in, const int* in_sizes, int n_in,
                              void* d_out, int out_size, void* d_ws, size_t ws_size,
                              hipStream_t stream) {
  const float* x     = (const float*)d_in[0];
  const float* W_in  = (const float*)d_in[1];
  const float* b_in  = (const float*)d_in[2];
  const float* W_out = (const float*)d_in[3];
  const float* b_out = (const float*)d_in[4];
  float* out = (float*)d_out;

  const size_t HSD = (size_t)NH * S_LEN * DH;
  unsigned short* Qb  = (unsigned short*)d_ws;       // [H,S,Dh] bf16 (q/8)
  unsigned short* Kb  = Qb + HSD;                    // [H,S,Dh] bf16
  unsigned short* Vt  = Kb + HSD;                    // [H,Dh,S] bf16
  unsigned short* Ob  = Vt + HSD;                    // [S,DM]   bf16
  unsigned short* Xb  = Ob + (size_t)S_LEN * DM;     // [S,DM]   bf16
  unsigned short* Wib = Xb + (size_t)S_LEN * DM;     // [3DM,DM] bf16
  unsigned short* Wob = Wib + (size_t)3 * DM * DM;   // [DM,DM]  bf16

  cvt_bf16_kernel<<<1536, 256, 0, stream>>>(x,     Xb,  (int)(S_LEN * DM / 8));
  cvt_bf16_kernel<<<864,  256, 0, stream>>>(W_in,  Wib, (int)(3 * DM * DM / 8));
  cvt_bf16_kernel<<<288,  256, 0, stream>>>(W_out, Wob, (int)(DM * DM / 8));

  qkv_gemm_kernel<<<2304, 128, 0, stream>>>(Xb, Wib, b_in, Qb, Kb, Vt);
  attn_kernel<<<768, 128, 0, stream>>>(Qb, Kb, Vt, Ob);
  out_gemm_kernel<<<768, 128, 0, stream>>>(Ob, Wob, b_out, out);
}
